// SelfAttention_21981642621436
// MI455X (gfx1250) — compile-verified, matrix operand order checked by reading, not run on hardware
//
#include <hip/hip_runtime.h>
#include <hip/hip_bf16.h>
#include <math.h>

typedef __attribute__((ext_vector_type(16))) __bf16 v16bf;
typedef __attribute__((ext_vector_type(8)))  __bf16 v8bf;
typedef __attribute__((ext_vector_type(8)))  float  v8f;

#define DIM_   1024
#define L_     2048
#define BATCH_ 4
#define NH_    16
#define HD_    64

#define KSTEP   64
#define APAD    8
#define ASTRIDE (KSTEP + APAD)   // 72 elements = 144B row stride: 16 lanes cover all 64 banks

__device__ __forceinline__ unsigned short f2bf(float f) {
    unsigned u = __builtin_bit_cast(unsigned, f);
    u += 0x7FFFu + ((u >> 16) & 1u);           // round-to-nearest-even
    return (unsigned short)(u >> 16);
}
__device__ __forceinline__ __bf16 f2bf16(float f) {
    unsigned short s = f2bf(f);
    return __builtin_bit_cast(__bf16, s);
}

// 16-bit A/B WMMA fragment from a row-major row pointer.
// e<8 -> K = kbase + 8*hl + e ; e>=8 -> K = kbase + 16 + 8*hl + (e-8)
__device__ __forceinline__ v16bf ldfrag(const __bf16* row, int kbase, int hl) {
    const int c = kbase + hl * 8;
    v8bf lo = *(const v8bf*)(row + c);
    v8bf hi = *(const v8bf*)(row + c + 16);
    return __builtin_shufflevector(lo, hi, 0,1,2,3,4,5,6,7,8,9,10,11,12,13,14,15);
}

__device__ __forceinline__ v8f wmma_bf(v16bf a, v16bf b, v8f c) {
    return __builtin_amdgcn_wmma_f32_16x16x32_bf16(false, a, false, b, (short)0, c, false, false);
}

// CDNA5 async global->LDS copy (ASYNCcnt-tracked), 16B per lane.
__device__ __forceinline__ void async_b128(const __bf16* gsrc, unsigned lds_off) {
    asm volatile("global_load_async_to_lds_b128 %0, %1, off"
                 :: "v"(lds_off), "v"(gsrc)
                 : "memory");
}
__device__ __forceinline__ void wait_async0() {
    asm volatile("s_wait_asynccnt 0" ::: "memory");
}

// ---------------------------------------------------------------- prep kernels
__global__ void __launch_bounds__(256) cvt_bf16_v8_kernel(const float* __restrict__ src,
                                                          __bf16* __restrict__ dst, int n8) {
    int i = blockIdx.x * 256 + threadIdx.x;
    if (i < n8) {
        const float4* s4 = (const float4*)src;
        float4 a = s4[2 * i], b = s4[2 * i + 1];
        v8bf o;
        o[0] = f2bf16(a.x); o[1] = f2bf16(a.y); o[2] = f2bf16(a.z); o[3] = f2bf16(a.w);
        o[4] = f2bf16(b.x); o[5] = f2bf16(b.y); o[6] = f2bf16(b.z); o[7] = f2bf16(b.w);
        *(v8bf*)(dst + 8 * i) = o;
    }
}

// dst[n*K + k] = bf16(src[k*N + n]) — LDS-tiled so both sides are coalesced.
__global__ void __launch_bounds__(256) transpose_bf16_kernel(const float* __restrict__ src,
                                                             __bf16* __restrict__ dst,
                                                             int K, int N) {
    __shared__ __bf16 tile[32][33];
    const int tx = threadIdx.x & 31;
    const int ty = threadIdx.x >> 5;
    const int k0 = blockIdx.y * 32, n0 = blockIdx.x * 32;
#pragma unroll
    for (int kk = ty; kk < 32; kk += 8)
        tile[tx][kk] = f2bf16(src[(size_t)(k0 + kk) * N + n0 + tx]);
    __syncthreads();
#pragma unroll
    for (int nn = ty; nn < 32; nn += 8)
        dst[(size_t)(n0 + nn) * K + k0 + tx] = tile[nn][tx];
}

// ---------------------------------------------------------------- LDS-tiled GEMM core
// Block: 256 threads = 8 waves, tile 128M x 128N, k-step 64, double-buffered async staging.
// Wave (wm,wn) computes 32M x 64N with 8 f32 accumulators (16 WMMAs per k-step).
__device__ __forceinline__ void gemm_stage(const __bf16* __restrict__ Ag,
                                           const __bf16* __restrict__ Bg,
                                           int m0, int n0, int K, int kk,
                                           __bf16* sA, __bf16* sB, int tid) {
#pragma unroll
    for (int it = 0; it < 4; ++it) {
        const int idx = tid + it * 256;     // 0..1023 : 128 rows x 8 chunks of 8 bf16
        const int row = idx >> 3;
        const int kc  = (idx & 7) * 8;
        async_b128(Ag + (size_t)(m0 + row) * K + kk + kc,
                   (unsigned)(size_t)(sA + row * ASTRIDE + kc));
        async_b128(Bg + (size_t)(n0 + row) * K + kk + kc,
                   (unsigned)(size_t)(sB + row * ASTRIDE + kc));
    }
}

__device__ __forceinline__ void gemm_compute_step(const __bf16* sA, const __bf16* sB,
                                                  int wm, int wn, int ln, int hl,
                                                  v8f acc[2][4]) {
#pragma unroll
    for (int kc = 0; kc < KSTEP; kc += 32) {
        v16bf a0 = ldfrag(sA + (wm * 32 + ln) * ASTRIDE, kc, hl);
        v16bf a1 = ldfrag(sA + (wm * 32 + 16 + ln) * ASTRIDE, kc, hl);
#pragma unroll
        for (int j = 0; j < 4; ++j) {
            v16bf b = ldfrag(sB + (wn * 64 + j * 16 + ln) * ASTRIDE, kc, hl);
            acc[0][j] = wmma_bf(a0, b, acc[0][j]);
            acc[1][j] = wmma_bf(a1, b, acc[1][j]);
        }
    }
}

// ---------------------------------------------------------------- QKV projection
__global__ void __launch_bounds__(256) qkv_gemm_kernel(const __bf16* __restrict__ xbf,
                                                       const __bf16* __restrict__ wT,
                                                       const float*  __restrict__ bqkv,
                                                       __bf16* __restrict__ Qm,
                                                       __bf16* __restrict__ Km,
                                                       __bf16* __restrict__ Vt) {
    __shared__ __bf16 sA[2][128 * ASTRIDE];
    __shared__ __bf16 sB[2][128 * ASTRIDE];
    const int tid = threadIdx.x, lane = tid & 31, wave = tid >> 5;
    const int ln = lane & 15, hl = lane >> 4;
    const int wm = wave >> 1, wn = wave & 1;
    const int m0 = blockIdx.y * 128, n0 = blockIdx.x * 128;
    v8f acc[2][4] = {};

    gemm_stage(xbf, wT, m0, n0, DIM_, 0, sA[0], sB[0], tid);
    wait_async0();
    __syncthreads();
    const int NK = DIM_ / KSTEP;
    for (int ks = 0; ks < NK; ++ks) {
        const int cur = ks & 1;
        if (ks + 1 < NK)
            gemm_stage(xbf, wT, m0, n0, DIM_, (ks + 1) * KSTEP, sA[cur ^ 1], sB[cur ^ 1], tid);
        gemm_compute_step(sA[cur], sB[cur], wm, wn, ln, hl, acc);
        wait_async0();
        __syncthreads();
    }
#pragma unroll
    for (int i = 0; i < 2; ++i)
#pragma unroll
    for (int j = 0; j < 4; ++j) {
        const int c = n0 + wn * 64 + j * 16 + ln;      // output column 0..3071
        const float bias = bqkv[c];
#pragma unroll
        for (int r = 0; r < 8; ++r) {
            const int m  = m0 + wm * 32 + i * 16 + 8 * hl + r;   // global row (b*L + pos)
            const int b  = m >> 11;
            const int lp = m & (L_ - 1);
            const __bf16 v = f2bf16(acc[i][j][r] + bias);
            if (c < DIM_) {
                const int hh = c >> 6, d = c & 63;
                Qm[((size_t)(b * NH_ + hh) * L_ + lp) * HD_ + d] = v;
            } else if (c < 2 * DIM_) {
                const int cc = c - DIM_;
                const int hh = cc >> 6, d = cc & 63;
                Km[((size_t)(b * NH_ + hh) * L_ + lp) * HD_ + d] = v;
            } else {
                const int cc = c - 2 * DIM_;
                const int hh = cc >> 6, d = cc & 63;
                Vt[((size_t)(b * NH_ + hh) * HD_ + d) * L_ + lp] = v;  // V transposed [d, L]
            }
        }
    }
}

// ---------------------------------------------------------------- flash attention
__constant__ const float SC_ = 0.125f * 1.44269504088896340736f;  // (1/sqrt(64)) * log2(e)

__device__ __forceinline__ v16bf softmax_update(const v8f& s0, const v8f& s1,
                                                int k0, int qq, int hl,
                                                float& mi, float& li, float& alpha) {
    float sl0[8], sl1[8];
    float mx = -3.0e38f;
#pragma unroll
    for (int r = 0; r < 8; ++r) {                  // acc VGPR r -> key = base + 8*hl + r
        const int key0 = k0 + 8 * hl + r;
        const int key1 = k0 + 16 + 8 * hl + r;
        sl0[r] = (key0 > qq) ? -3.0e38f : s0[r] * SC_;
        sl1[r] = (key1 > qq) ? -3.0e38f : s1[r] * SC_;
        mx = fmaxf(mx, fmaxf(sl0[r], sl1[r]));
    }
    mx = fmaxf(mx, __shfl_xor(mx, 16, 32));        // combine both key halves of this query
    const float mn = fmaxf(mi, mx);
    alpha = exp2f(mi - mn);
    float ps = 0.0f;
    v16bf pb;
#pragma unroll
    for (int r = 0; r < 8; ++r) {
        const float p0 = exp2f(sl0[r] - mn);
        const float p1 = exp2f(sl1[r] - mn);
        ps += p0 + p1;
        pb[r]     = f2bf16(p0);                    // P^T B-fragment straight from S^T accs
        pb[r + 8] = f2bf16(p1);
    }
    ps += __shfl_xor(ps, 16, 32);
    li = li * alpha + ps;
    mi = mn;
    return pb;
}

// One wave handles 32 queries (two 16-q tiles) of one (b,h).
// Software-pipelined: K fragments preloaded one 32-key block ahead; V fragments
// issued before the softmax VALU so their latency hides under v_exp work.
__global__ void __launch_bounds__(32) fa_kernel(const __bf16* __restrict__ Qm,
                                                const __bf16* __restrict__ Km,
                                                const __bf16* __restrict__ Vt,
                                                __bf16* __restrict__ Obuf) {
    const int lane = threadIdx.x;
    const int ln = lane & 15, hl = lane >> 4;
    const int bh = blockIdx.y;
    const int b = bh >> 4, h = bh & 15;
    const __bf16* Qb = Qm + (size_t)bh * L_ * HD_;
    const __bf16* Kb = Km + (size_t)bh * L_ * HD_;
    const __bf16* Vb = Vt + (size_t)bh * HD_ * L_;

    const int q0 = blockIdx.x * 32;
    const int qqA = q0 + ln, qqB = q0 + 16 + ln;
    const __bf16* qrA = Qb + (size_t)(q0 + ln) * HD_;
    const __bf16* qrB = Qb + (size_t)(q0 + 16 + ln) * HD_;
    v16bf bqA0 = ldfrag(qrA, 0, hl), bqA1 = ldfrag(qrA, 32, hl);
    v16bf bqB0 = ldfrag(qrB, 0, hl), bqB1 = ldfrag(qrB, 32, hl);

    v8f oA[4] = {}, oB[4] = {};
    float miA = -3.0e38f, liA = 0.0f, miB = -3.0e38f, liB = 0.0f;

    const int nkb = (q0 >> 5) + 1;                 // 32-key blocks up to causal edge
    // prologue: preload K fragments for block 0
    v16bf kf00 = ldfrag(Kb + (size_t)(ln) * HD_, 0, hl);
    v16bf kf01 = ldfrag(Kb + (size_t)(ln) * HD_, 32, hl);
    v16bf kf10 = ldfrag(Kb + (size_t)(16 + ln) * HD_, 0, hl);
    v16bf kf11 = ldfrag(Kb + (size_t)(16 + ln) * HD_, 32, hl);

    for (int kb = 0; kb < nkb; ++kb) {
        const int k0 = kb * 32;
        v8f sA0 = {}, sA1 = {}, sB0 = {}, sB1 = {};
        sA0 = wmma_bf(kf00, bqA0, sA0); sA0 = wmma_bf(kf01, bqA1, sA0);
        sA1 = wmma_bf(kf10, bqA0, sA1); sA1 = wmma_bf(kf11, bqA1, sA1);
        sB0 = wmma_bf(kf00, bqB0, sB0); sB0 = wmma_bf(kf01, bqB1, sB0);
        sB1 = wmma_bf(kf10, bqB0, sB1); sB1 = wmma_bf(kf11, bqB1, sB1);

        // V fragments for this block (consumed after softmax)
        v16bf vf0 = ldfrag(Vb + (size_t)( 0 + ln) * L_, k0, hl);
        v16bf vf1 = ldfrag(Vb + (size_t)(16 + ln) * L_, k0, hl);
        v16bf vf2 = ldfrag(Vb + (size_t)(32 + ln) * L_, k0, hl);
        v16bf vf3 = ldfrag(Vb + (size_t)(48 + ln) * L_, k0, hl);
        // prefetch next block's K fragments (clamped on last iter; control flow uniform)
        const int k1 = (kb + 1 < nkb) ? (k0 + 32) : k0;
        const __bf16* nr0 = Kb + (size_t)(k1 + ln) * HD_;
        const __bf16* nr1 = Kb + (size_t)(k1 + 16 + ln) * HD_;
        v16bf nk00 = ldfrag(nr0, 0, hl), nk01 = ldfrag(nr0, 32, hl);
        v16bf nk10 = ldfrag(nr1, 0, hl), nk11 = ldfrag(nr1, 32, hl);

        float aA, aB;
        v16bf pbA = softmax_update(sA0, sA1, k0, qqA, hl, miA, liA, aA);
        v16bf pbB = softmax_update(sB0, sB1, k0, qqB, hl, miB, liB, aB);
#pragma unroll
        for (int e = 0; e < 8; ++e) {
            oA[0][e] *= aA; oA[1][e] *= aA; oA[2][e] *= aA; oA[3][e] *= aA;
            oB[0][e] *= aB; oB[1][e] *= aB; oB[2][e] *= aB; oB[3][e] *= aB;
        }
        oA[0] = wmma_bf(vf0, pbA, oA[0]); oB[0] = wmma_bf(vf0, pbB, oB[0]);
        oA[1] = wmma_bf(vf1, pbA, oA[1]); oB[1] = wmma_bf(vf1, pbB, oB[1]);
        oA[2] = wmma_bf(vf2, pbA, oA[2]); oB[2] = wmma_bf(vf2, pbB, oB[2]);
        oA[3] = wmma_bf(vf3, pbA, oA[3]); oB[3] = wmma_bf(vf3, pbB, oB[3]);

        kf00 = nk00; kf01 = nk01; kf10 = nk10; kf11 = nk11;
    }

    const float ivA = 1.0f / liA, ivB = 1.0f / liB;
    __bf16* orA = Obuf + (size_t)(b * L_ + q0 + ln) * DIM_ + h * HD_;
    __bf16* orB = Obuf + (size_t)(b * L_ + q0 + 16 + ln) * DIM_ + h * HD_;
#pragma unroll
    for (int j = 0; j < 4; ++j)
#pragma unroll
    for (int r = 0; r < 8; ++r) {                  // O^T: lane = query, d = j*16 + 8*hl + r
        orA[j * 16 + 8 * hl + r] = f2bf16(oA[j][r] * ivA);
        orB[j * 16 + 8 * hl + r] = f2bf16(oB[j][r] * ivB);
    }
}

// ---------------------------------------------------------------- output projection
__global__ void __launch_bounds__(256) out_gemm_kernel(const __bf16* __restrict__ obf,
                                                       const __bf16* __restrict__ wT,
                                                       const float*  __restrict__ bout,
                                                       float* __restrict__ out) {
    __shared__ __bf16 sA[2][128 * ASTRIDE];
    __shared__ __bf16 sB[2][128 * ASTRIDE];
    const int tid = threadIdx.x, lane = tid & 31, wave = tid >> 5;
    const int ln = lane & 15, hl = lane >> 4;
    const int wm = wave >> 1, wn = wave & 1;
    const int m0 = blockIdx.y * 128, n0 = blockIdx.x * 128;
    v8f acc[2][4] = {};

    gemm_stage(obf, wT, m0, n0, DIM_, 0, sA[0], sB[0], tid);
    wait_async0();
    __syncthreads();
    const int NK = DIM_ / KSTEP;
    for (int ks = 0; ks < NK; ++ks) {
        const int cur = ks & 1;
        if (ks + 1 < NK)
            gemm_stage(obf, wT, m0, n0, DIM_, (ks + 1) * KSTEP, sA[cur ^ 1], sB[cur ^ 1], tid);
        gemm_compute_step(sA[cur], sB[cur], wm, wn, ln, hl, acc);
        wait_async0();
        __syncthreads();
    }
#pragma unroll
    for (int i = 0; i < 2; ++i)
#pragma unroll
    for (int j = 0; j < 4; ++j) {
        const int c = n0 + wn * 64 + j * 16 + ln;
        const float bias = bout[c];
#pragma unroll
        for (int r = 0; r < 8; ++r) {
            const int m = m0 + wm * 32 + i * 16 + 8 * hl + r;
            out[(size_t)m * DIM_ + c] = acc[i][j][r] + bias;
        }
    }
}

// ---------------------------------------------------------------- launcher
extern "C" void kernel_launch(void* const* d_in, const int* in_sizes, int n_in,
                              void* d_out, int out_size, void* d_ws, size_t ws_size,
                              hipStream_t stream) {
    (void)in_sizes; (void)n_in; (void)out_size; (void)ws_size;
    const float* x    = (const float*)d_in[0];   // [4,2048,1024]
    const float* wqkv = (const float*)d_in[1];   // [1024,3072]
    const float* bqkv = (const float*)d_in[2];   // [3072]
    const float* wout = (const float*)d_in[3];   // [1024,1024]
    const float* bout = (const float*)d_in[4];   // [1024]

    char* w = (char*)d_ws;
    const size_t MB = 1u << 20;
    __bf16* xbf   = (__bf16*)(w + 0 * MB);    // 16 MB  (reused as Obuf later)
    __bf16* wqkvT = (__bf16*)(w + 16 * MB);   //  6 MB  [3072,1024]
    __bf16* woutT = (__bf16*)(w + 22 * MB);   //  2 MB  [1024,1024]
    __bf16* Qm    = (__bf16*)(w + 24 * MB);   // 16 MB  [B,H,L,64]
    __bf16* Km    = (__bf16*)(w + 40 * MB);   // 16 MB  [B,H,L,64]
    __bf16* Vt    = (__bf16*)(w + 56 * MB);   // 16 MB  [B,H,64,L]
    __bf16* Obuf  = xbf;                      // xbf dead after qkv_gemm

    const int nx = BATCH_ * L_ * DIM_;        // 8388608
    cvt_bf16_v8_kernel<<<(nx / 8 + 255) / 256, 256, 0, stream>>>(x, xbf, nx / 8);
    transpose_bf16_kernel<<<dim3(3 * DIM_ / 32, DIM_ / 32), 256, 0, stream>>>(wqkv, wqkvT, DIM_, 3 * DIM_);
    transpose_bf16_kernel<<<dim3(DIM_ / 32, DIM_ / 32), 256, 0, stream>>>(wout, woutT, DIM_, DIM_);

    qkv_gemm_kernel<<<dim3(3 * DIM_ / 128, BATCH_ * L_ / 128), 256, 0, stream>>>(xbf, wqkvT, bqkv, Qm, Km, Vt);
    fa_kernel<<<dim3(L_ / 32, BATCH_ * NH_), 32, 0, stream>>>(Qm, Km, Vt, Obuf);
    out_gemm_kernel<<<dim3(DIM_ / 128, BATCH_ * L_ / 128), 256, 0, stream>>>(Obuf, woutT, bout, (float*)d_out);
}
